// SurvNODE_11708080849252
// MI455X (gfx1250) — matrix operand order repaired and checked by reading, not compiled, not run on hardware
//
#include <hip/hip_runtime.h>
#include <hip/hip_bf16.h>
#include <math.h>

typedef __attribute__((ext_vector_type(16))) _Float16 v16h;
typedef __attribute__((ext_vector_type(8)))  _Float16 v8h;
typedef __attribute__((ext_vector_type(2)))  _Float16 v2h;
typedef __attribute__((ext_vector_type(8)))  float    v8f;

#define NSUB 8192
#define GSTEP 256
#define KPAD 96      // 86 -> padded to 3 K-tiles of 32
#define HID 128
#define PAIRS_PER_BLK 2
#define SUBJ_PER_BLK (PAIRS_PER_BLK * 16)   // 32 subjects / block, 256 blocks

// ---------- helpers ----------

#if __has_builtin(__builtin_amdgcn_tanhf)
__device__ __forceinline__ float fast_tanhf(float x) { return __builtin_amdgcn_tanhf(x); }
#else
__device__ __forceinline__ float fast_tanhf(float x) { return tanhf(x); }
#endif

__device__ __forceinline__ v16h load_afrag(const _Float16* p) {
    // lane's A fragment: 8 halfs at p, 8 halfs at p+16 (K..K+7, K+16..K+23)
    v8h lo = *(const v8h*)(p);
    v8h hi = *(const v8h*)(p + 16);
    return __builtin_shufflevector(lo, hi, 0,1,2,3,4,5,6,7,8,9,10,11,12,13,14,15);
}

__device__ __forceinline__ v16h load_bfrag(const _Float16* p) {
    // lane's B fragment: 8 halfs at p, 8 halfs at p+16 (K..K+7, K+16..K+23)
    v8h lo = *(const v8h*)(p);
    v8h hi = *(const v8h*)(p + 16);
    return __builtin_shufflevector(lo, hi, 0,1,2,3,4,5,6,7,8,9,10,11,12,13,14,15);
}

__device__ __forceinline__ float softplusf(float v) {
    return v > 0.f ? v + log1pf(expf(-v)) : log1pf(expf(v));
}

// Layer 1, N-split across a wave pair: wave role r handles N-tiles r*4 .. r*4+3.
__device__ __forceinline__ void run_l1(const _Float16* __restrict__ sW1, // [128][96] col-major
                                       const float* __restrict__ sB1,
                                       const _Float16* __restrict__ sH,   // [16][96]
                                       _Float16* __restrict__ sHid,       // [16][128]
                                       int lane, int r) {
    const int m  = lane & 15;
    const int hi = lane >> 4;
    v16h A[3];
#pragma unroll
    for (int kt = 0; kt < 3; ++kt)
        A[kt] = load_afrag(sH + m * KPAD + kt * 32 + hi * 8);

#pragma unroll
    for (int nt4 = 0; nt4 < 4; ++nt4) {
        const int nt = r * 4 + nt4;
        const int n  = nt * 16 + m;
        v8f c = {};
#pragma unroll
        for (int kt = 0; kt < 3; ++kt) {
            v16h b = load_bfrag(sW1 + n * KPAD + kt * 32 + hi * 8);
            c = __builtin_amdgcn_wmma_f32_16x16x32_f16(false, A[kt], false, b,
                                                       (short)0, c, false, false);
        }
        const float bn = sB1[n];
#pragma unroll
        for (int j = 0; j < 8; ++j) {
            float v = fast_tanhf(c[j] + bn);
            sHid[(j + hi * 8) * HID + n] = (_Float16)v;  // row = subject, col = hidden n
        }
    }
}

// Layer 2 (wave 0 of each pair only): [16 x 128] @ [128 x 16]
__device__ __forceinline__ void run_l2(const _Float16* __restrict__ sW2, // [16][128] col-major
                                       const float* __restrict__ sB2,
                                       const _Float16* __restrict__ sHid,
                                       float* __restrict__ sOut,          // [16][16]
                                       int lane) {
    const int m  = lane & 15;
    const int hi = lane >> 4;
    const int n  = m;
    v8f c = {};
#pragma unroll
    for (int kt = 0; kt < 4; ++kt) {
        v16h a = load_afrag(sHid + m * HID + kt * 32 + hi * 8);
        v16h b = load_bfrag(sW2 + n * HID + kt * 32 + hi * 8);
        c = __builtin_amdgcn_wmma_f32_16x16x32_f16(false, a, false, b,
                                                   (short)0, c, false, false);
    }
    const float bn = sB2[n];
#pragma unroll
    for (int j = 0; j < 8; ++j)
        sOut[(j + hi * 8) * 16 + n] = c[j] + bn;
}

// Pack the 21-float state (+nothing else) into h[0..20] as f16 using b32 pair stores.
__device__ __forceinline__ void write_state_h(_Float16* hrow, const float* st) {
#pragma unroll
    for (int j = 0; j < 10; ++j) {
        v2h p;
        p[0] = (_Float16)st[2 * j];
        p[1] = (_Float16)st[2 * j + 1];
        *(v2h*)(hrow + 2 * j) = p;
    }
    hrow[20] = (_Float16)st[20];
}

// ---------- kernels ----------

__global__ void tmax_kernel(const float* __restrict__ tstop, float* __restrict__ outmax, int n) {
    __shared__ float red[256];
    float mx = -1e30f;
    for (int i = threadIdx.x; i < n; i += 256) mx = fmaxf(mx, tstop[i]);
    red[threadIdx.x] = mx;
    __syncthreads();
    for (int s = 128; s > 0; s >>= 1) {
        if (threadIdx.x < (unsigned)s) red[threadIdx.x] = fmaxf(red[threadIdx.x], red[threadIdx.x + s]);
        __syncthreads();
    }
    if (threadIdx.x == 0) *outmax = red[0];
}

__global__ void enc_kernel(const float* __restrict__ x,
                           const float* __restrict__ W1, const float* __restrict__ b1,
                           const float* __restrict__ W2, const float* __restrict__ b2,
                           float* __restrict__ z, int n) {
    int i = blockIdx.x * blockDim.x + threadIdx.x;
    if (i >= n) return;
    float xi[32];
#pragma unroll
    for (int j = 0; j < 32; ++j) xi[j] = x[i * 32 + j];
    float h[64];
#pragma unroll 4
    for (int j = 0; j < 64; ++j) {
        float a = b1[j];
#pragma unroll
        for (int k = 0; k < 32; ++k) a += xi[k] * W1[k * 64 + j];
        h[j] = fast_tanhf(a);
    }
#pragma unroll 4
    for (int o = 0; o < 32; ++o) {
        float a = b2[o];
#pragma unroll
        for (int j = 0; j < 64; ++j) a += h[j] * W2[j * 32 + o];
        z[i * 32 + o] = a;
    }
}

__global__ __launch_bounds__(128) void surv_main(
    const float* __restrict__ x, const float* __restrict__ tstop,
    const int* __restrict__ from_state, const int* __restrict__ to_state,
    const int* __restrict__ tstart_idx, const int* __restrict__ tstop_idx,
    const float* __restrict__ times,
    const float* __restrict__ net_W1, const float* __restrict__ net_b1,
    const float* __restrict__ net_W2, const float* __restrict__ net_b2,
    const float* __restrict__ zbuf, const float* __restrict__ tmaxp,
    float* __restrict__ out) {

    // ---- LDS ----
    __shared__ __align__(16) _Float16 sW1[HID * KPAD];   // 24 KB, col-major [n][k]
    __shared__ __align__(16) _Float16 sW2[16 * HID];     // 4 KB,  col-major [n][k]
    __shared__ float  sB1[HID];
    __shared__ float  sB2[16];
    __shared__ float  sTimes[GSTEP + 1];
    __shared__ float  sTmax;
    __shared__ __align__(16) _Float16 sH[PAIRS_PER_BLK * 16 * KPAD];   // 6 KB
    __shared__ __align__(16) _Float16 sHid[PAIRS_PER_BLK * 16 * HID];  // 8 KB
    __shared__ float  sOut2[PAIRS_PER_BLK * 16 * 16];                  // 2 KB

    const int tid  = threadIdx.x;
    const int w    = tid >> 5;
    const int pair = w >> 1;
    const int r    = w & 1;      // role within pair: 0 = leader
    const int lane = tid & 31;
    const int m    = lane & 15;

    __builtin_prefetch(net_W1, 0, 0);

    // ---- stage weights (f16) into LDS ----
    for (int i = tid; i < HID * KPAD; i += 128) {
        int n = i / KPAD, k = i % KPAD;
        sW1[i] = (_Float16)((k < 86) ? net_W1[k * HID + n] : 0.f);
    }
    for (int i = tid; i < 16 * HID; i += 128) {
        int n = i / HID, k = i % HID;
        sW2[i] = (_Float16)((n < 6) ? net_W2[k * 6 + n] : 0.f);
    }
    for (int i = tid; i < HID; i += 128) sB1[i] = net_b1[i];
    if (tid < 16) sB2[tid] = (tid < 6) ? net_b2[tid] : 0.f;
    for (int i = tid; i <= GSTEP; i += 128) sTimes[i] = times[i];
    if (tid == 0) sTmax = *tmaxp;

    _Float16* myH   = sH    + pair * 16 * KPAD;
    _Float16* myHid = sHid  + pair * 16 * HID;
    float*    myOut = sOut2 + pair * 16 * 16;

    const int subj = blockIdx.x * SUBJ_PER_BLK + pair * 16 + m;

    // ---- per-lane persistent state (wave 0 of each pair, lanes 0..15) ----
    float P[9], Ti[9], qi[3];
    float stopS[21], tinvStart[9];
    int   tsi = 0, tei = 0;

    if (r == 0 && lane < 16) {
        _Float16* hrow = myH + m * KPAD;
#pragma unroll
        for (int j = 0; j < 32; ++j) hrow[21 + j] = (_Float16)zbuf[subj * 32 + j];
#pragma unroll
        for (int j = 0; j < 32; ++j) hrow[53 + j] = (_Float16)x[subj * 32 + j];
#pragma unroll
        for (int j = 86; j < KPAD; ++j) hrow[j] = (_Float16)0.f;
#pragma unroll
        for (int j = 0; j < 9; ++j) { P[j] = (j % 4 == 0) ? 1.f : 0.f; Ti[j] = P[j]; }
#pragma unroll
        for (int j = 0; j < 3; ++j) qi[j] = 0.f;
        float st0[21];
#pragma unroll
        for (int j = 0; j < 9; ++j) { st0[j] = P[j]; st0[9 + j] = Ti[j]; }
#pragma unroll
        for (int j = 0; j < 3; ++j) st0[18 + j] = 0.f;
        write_state_h(hrow, st0);
        hrow[85] = (_Float16)0.f;   // times[0]

        tsi = tstart_idx[subj];
        tei = tstop_idx[subj];
#pragma unroll
        for (int j = 0; j < 9; ++j) tinvStart[j] = Ti[j];    // valid if tsi == 0
#pragma unroll
        for (int j = 0; j < 21; ++j) stopS[j] = st0[j];      // valid if tei == 0
    }
    __syncthreads();

    // ---- 256 Euler steps ----
#pragma unroll 1
    for (int s = 0; s < GSTEP; ++s) {
        run_l1(sW1, sB1, myH, myHid, lane, r);   // both waves of the pair
        __syncthreads();                          // sHid complete

        if (r == 0) {
            run_l2(sW2, sB2, myHid, myOut, lane);
            if (lane < 16) {
                const float t  = sTimes[s];
                const float dt = sTimes[s + 1] - t;
                float q0 = softplusf(myOut[m * 16 + 0]);
                float q1 = softplusf(myOut[m * 16 + 1]);
                float q2 = softplusf(myOut[m * 16 + 2]);
                float Q[9] = { -(q0 + q1), q0,  q1,
                                0.f,      -q2,  q2,
                                0.f,       0.f, 0.f };
                float nP[9], nT[9];
#pragma unroll
                for (int i = 0; i < 3; ++i)
#pragma unroll
                    for (int k = 0; k < 3; ++k) {
                        float dp = 0.f, dti = 0.f;
#pragma unroll
                        for (int j = 0; j < 3; ++j) {
                            dp  += P[i * 3 + j] * Q[j * 3 + k];
                            dti -= Q[i * 3 + j] * Ti[j * 3 + k];
                        }
                        nP[i * 3 + k] = P[i * 3 + k]  + dt * dp;
                        nT[i * 3 + k] = Ti[i * 3 + k] + dt * dti;
                    }
#pragma unroll
                for (int j = 0; j < 9; ++j) { P[j] = nP[j]; Ti[j] = nT[j]; }
                qi[0] += dt * q0; qi[1] += dt * q1; qi[2] += dt * q2;

                const int idx = s + 1;
                if (idx == tsi) {
#pragma unroll
                    for (int j = 0; j < 9; ++j) tinvStart[j] = Ti[j];
                }
                float st[21];
#pragma unroll
                for (int j = 0; j < 9; ++j) { st[j] = P[j]; st[9 + j] = Ti[j]; }
#pragma unroll
                for (int j = 0; j < 3; ++j) st[18 + j] = qi[j];
                if (idx == tei) {
#pragma unroll
                    for (int j = 0; j < 21; ++j) stopS[j] = st[j];
                }
                _Float16* hrow = myH + m * KPAD;
                write_state_h(hrow, st);
                hrow[85] = (_Float16)sTimes[s + 1];
            }
        }
        __syncthreads();                          // sH updated; sHid free for reuse
    }

    // ---- eval at (state_stop, tstop): lam and S ----
    if (r == 0 && lane < 16) {
        _Float16* hrow = myH + m * KPAD;
        write_state_h(hrow, stopS);
        hrow[85] = (_Float16)tstop[subj];
    }
    __syncthreads();
    run_l1(sW1, sB1, myH, myHid, lane, r);
    __syncthreads();
    if (r == 0) {
        run_l2(sW2, sB2, myHid, myOut, lane);
        if (lane < 16) {
            float qv[3];
#pragma unroll
            for (int k = 0; k < 3; ++k) qv[k] = softplusf(myOut[m * 16 + k]);
            const int fi = from_state[subj] - 1;
            const int ti = to_state[subj] - 1;
            const int k  = (fi == 0) ? ((ti == 1) ? 0 : 1) : 2;  // (0,1)(0,2)(1,2)
            float S = 0.f;
#pragma unroll
            for (int j = 0; j < 3; ++j) S += tinvStart[fi * 3 + j] * stopS[j * 3 + fi];
            out[subj]        = S;
            out[NSUB + subj] = qv[k];

            // stage h for hazards eval: final state + tmax
            float st[21];
#pragma unroll
            for (int j = 0; j < 9; ++j) { st[j] = P[j]; st[9 + j] = Ti[j]; }
#pragma unroll
            for (int j = 0; j < 3; ++j) st[18 + j] = qi[j];
            _Float16* hrow = myH + m * KPAD;
            write_state_h(hrow, st);
            hrow[85] = (_Float16)sTmax;
        }
    }
    __syncthreads();

    // ---- eval at (final state, max(tstop)): hazards ----
    run_l1(sW1, sB1, myH, myHid, lane, r);
    __syncthreads();
    if (r == 0) {
        run_l2(sW2, sB2, myHid, myOut, lane);
        if (lane < 16) {
#pragma unroll
            for (int j = 0; j < 3; ++j)
                out[2 * NSUB + subj * 3 + j] = myOut[m * 16 + 3 + j];
        }
    }
}

// ---------- launcher ----------

extern "C" void kernel_launch(void* const* d_in, const int* in_sizes, int n_in,
                              void* d_out, int out_size, void* d_ws, size_t ws_size,
                              hipStream_t stream) {
    const float* x          = (const float*)d_in[0];
    const float* tstop      = (const float*)d_in[2];
    const int*   from_state = (const int*)d_in[3];
    const int*   to_state   = (const int*)d_in[4];
    const int*   tstart_idx = (const int*)d_in[5];
    const int*   tstop_idx  = (const int*)d_in[6];
    const float* times      = (const float*)d_in[7];
    const float* enc_W1     = (const float*)d_in[10];
    const float* enc_b1     = (const float*)d_in[11];
    const float* enc_W2     = (const float*)d_in[12];
    const float* enc_b2     = (const float*)d_in[13];
    const float* net_W1     = (const float*)d_in[14];
    const float* net_b1     = (const float*)d_in[15];
    const float* net_W2     = (const float*)d_in[16];
    const float* net_b2     = (const float*)d_in[17];

    float* out   = (float*)d_out;
    float* zbuf  = (float*)d_ws;            // N*32 floats of encoder output
    float* tmaxp = zbuf + NSUB * 32;        // 1 float: max(tstop)

    tmax_kernel<<<1, 256, 0, stream>>>(tstop, tmaxp, NSUB);
    enc_kernel<<<NSUB / 256, 256, 0, stream>>>(x, enc_W1, enc_b1, enc_W2, enc_b2, zbuf, NSUB);
    surv_main<<<NSUB / SUBJ_PER_BLK, 128, 0, stream>>>(
        x, tstop, from_state, to_state, tstart_idx, tstop_idx, times,
        net_W1, net_b1, net_W2, net_b2, zbuf, tmaxp, out);
}
